// GAT_1022202217424
// MI455X (gfx1250) — matrix operand order repaired and checked by reading, not compiled, not run on hardware
//
#include <hip/hip_runtime.h>
#include <hip/hip_fp16.h>

#define NNODES 50000
#define NEDGES 400000

typedef __attribute__((ext_vector_type(16))) _Float16 v16h;
typedef __attribute__((ext_vector_type(8)))  float    v8f;

union FragU { uint4 u[2]; v16h h; };

static inline int divup_h(int a, int b) { return (a + b - 1) / b; }

// ---------------------------------------------------------------- utilities
__global__ void fill_f32(float* __restrict__ p, float v, int n) {
  int i = blockIdx.x * blockDim.x + threadIdx.x;
  if (i < n) p[i] = v;
}

__global__ void conv_f32_to_f16(const float* __restrict__ in, _Float16* __restrict__ out, int n) {
  int i = blockIdx.x * blockDim.x + threadIdx.x;
  if (i < n) out[i] = (_Float16)in[i];
}

// W [K, Nout] fp32 row-major -> Wt [Nout, K] f16 (so B fragments are contiguous)
__global__ void conv_w_transpose_f16(const float* __restrict__ W, _Float16* __restrict__ Wt,
                                     int K, int Nout) {
  int i = blockIdx.x * blockDim.x + threadIdx.x;
  if (i < K * Nout) {
    int k = i / Nout, n = i % Nout;
    Wt[(size_t)n * K + k] = (_Float16)W[i];
  }
}

// ---------------------------------------------------------------- WMMA GEMM
// D[M,Nout] = A[M,K] (f16) x Bt[Nout,K]^T (f16), f32 accumulate.
// One wave computes a 16x64 output strip: one A fragment reused across 4
// independent accumulators (breaks the WMMA C-operand RAW chain, 2.5 b128
// loads per v_wmma). Fragments packed in contiguous k order (k = 16*hi + s):
//   A lane(l): m = l&15, hi = l>>4; halves = A[m][16*hi + 0..7], A[m][16*hi + 8..15]
//   B lane(l): n = l&15, hi = l>>4; halves = B[16*hi + 0..15][n]  (contig in Bt)
//   C/D: vgpr r, lane l -> M = r + 8*(l>>4), N = l&15
__global__ void wmma_gemm_f16(const _Float16* __restrict__ A, const _Float16* __restrict__ Bt,
                              float* __restrict__ D, int K, int Nout,
                              int stripsN, int totalStrips) {
  int lane = threadIdx.x & 31;
  int wave = (blockIdx.x * blockDim.x + threadIdx.x) >> 5;
  if (wave >= totalStrips) return;  // uniform per-wave: EXEC stays all-ones
  int tm = wave / stripsN;
  int tn = wave % stripsN;
  int hi = lane >> 4;
  int nl = lane & 15;

  const _Float16* Arow = A + (size_t)(tm * 16 + nl) * K + 16 * hi;
  const _Float16* Brow[4];
#pragma unroll
  for (int j = 0; j < 4; ++j)
    Brow[j] = Bt + (size_t)(tn * 64 + j * 16 + nl) * K + 16 * hi;

  v8f c0 = {}, c1 = {}, c2 = {}, c3 = {};
  for (int k0 = 0; k0 < K; k0 += 32) {
    __builtin_prefetch(Arow + k0 + 64, 0, 0);  // global_prefetch_b8, speculative
    FragU fa;
    fa.u[0] = *reinterpret_cast<const uint4*>(Arow + k0);       // K = 16*hi + 0..7
    fa.u[1] = *reinterpret_cast<const uint4*>(Arow + k0 + 8);   // K = 16*hi + 8..15
    FragU fb0, fb1, fb2, fb3;
    fb0.u[0] = *reinterpret_cast<const uint4*>(Brow[0] + k0);
    fb0.u[1] = *reinterpret_cast<const uint4*>(Brow[0] + k0 + 8);
    fb1.u[0] = *reinterpret_cast<const uint4*>(Brow[1] + k0);
    fb1.u[1] = *reinterpret_cast<const uint4*>(Brow[1] + k0 + 8);
    fb2.u[0] = *reinterpret_cast<const uint4*>(Brow[2] + k0);
    fb2.u[1] = *reinterpret_cast<const uint4*>(Brow[2] + k0 + 8);
    fb3.u[0] = *reinterpret_cast<const uint4*>(Brow[3] + k0);
    fb3.u[1] = *reinterpret_cast<const uint4*>(Brow[3] + k0 + 8);
    c0 = __builtin_amdgcn_wmma_f32_16x16x32_f16(false, fa.h, false, fb0.h, (short)0, c0, false, false);
    c1 = __builtin_amdgcn_wmma_f32_16x16x32_f16(false, fa.h, false, fb1.h, (short)0, c1, false, false);
    c2 = __builtin_amdgcn_wmma_f32_16x16x32_f16(false, fa.h, false, fb2.h, (short)0, c2, false, false);
    c3 = __builtin_amdgcn_wmma_f32_16x16x32_f16(false, fa.h, false, fb3.h, (short)0, c3, false, false);
  }

  int mbase = tm * 16 + 8 * hi;
  size_t colbase = (size_t)tn * 64 + nl;
#pragma unroll
  for (int r = 0; r < 8; ++r) {
    float* drow = D + (size_t)(mbase + r) * Nout + colbase;
    drow[0]  = c0[r];
    drow[16] = c1[r];
    drow[32] = c2[r];
    drow[48] = c3[r];
  }
}

// ------------------------------------------------------- attention coefficients
// One wave per (node, head): alpha_s/d[n,h] = dot(h[n,h,:], a_src/dst[h,:])
__global__ void alpha_kernel(const float* __restrict__ h, const float* __restrict__ a_src,
                             const float* __restrict__ a_dst,
                             float* __restrict__ as, float* __restrict__ ad,
                             int Hh, int Cc) {
  int lane = threadIdx.x & 31;
  int wave = (blockIdx.x * blockDim.x + threadIdx.x) >> 5;
  if (wave >= NNODES * Hh) return;
  int n  = wave / Hh;
  int hd = wave % Hh;
  const float* hrow = h + ((size_t)n * Hh + hd) * Cc;
  const float* asr  = a_src + (size_t)hd * Cc;
  const float* adr  = a_dst + (size_t)hd * Cc;
  float ss = 0.f, sd = 0.f;
  for (int c = lane; c < Cc; c += 32) {
    float v = hrow[c];
    ss += v * asr[c];
    sd += v * adr[c];
  }
  for (int off = 16; off > 0; off >>= 1) {
    ss += __shfl_down(ss, off, 32);
    sd += __shfl_down(sd, off, 32);
  }
  if (lane == 0) { as[wave] = ss; ad[wave] = sd; }
}

// ---------------------------------------------------------------- edge passes
__global__ void edge_logits_max(const float* __restrict__ as, const float* __restrict__ ad,
                                const long long* __restrict__ src, const long long* __restrict__ dst,
                                float* __restrict__ e_ws, float* __restrict__ m, int Hh) {
  int i = blockIdx.x * blockDim.x + threadIdx.x;
  if (i >= NEDGES * Hh) return;
  int e  = i / Hh;
  int hd = i % Hh;
  int s = (int)src[e];
  int d = (int)dst[e];
  float v = as[s * Hh + hd] + ad[d * Hh + hd];
  v = v > 0.f ? v : 0.2f * v;  // leaky_relu
  e_ws[i] = v;
  atomicMax(&m[d * Hh + hd], v);  // global_atomic_max_num_f32
}

__global__ void edge_exp_sum(const float* __restrict__ e_ws, const float* __restrict__ m,
                             const long long* __restrict__ dst,
                             float* __restrict__ p_ws, float* __restrict__ den, int Hh) {
  int i = blockIdx.x * blockDim.x + threadIdx.x;
  if (i >= NEDGES * Hh) return;
  int e  = i / Hh;
  int hd = i % Hh;
  int d = (int)dst[e];
  float p = expf(e_ws[i] - m[d * Hh + hd]);
  p_ws[i] = p;
  atomicAdd(&den[d * Hh + hd], p);
}

// One wave per (edge, head): scatter coef * h[src,h,:] into acc[dst,h,:].
// b128 gather per lane (Cc == 128 -> 32 lanes x float4), per-dword f32 atomics.
__global__ void edge_scatter(const float* __restrict__ p_ws, const float* __restrict__ den,
                             const float* __restrict__ h,
                             const long long* __restrict__ src, const long long* __restrict__ dst,
                             float* __restrict__ acc, int Hh, int Cc) {
  int lane = threadIdx.x & 31;
  int wave = (blockIdx.x * blockDim.x + threadIdx.x) >> 5;
  if (wave >= NEDGES * Hh) return;
  int e  = wave / Hh;
  int hd = wave % Hh;
  int s = (int)src[e];
  int d = (int)dst[e];
  float coef = p_ws[wave] / (den[d * Hh + hd] + 1e-16f);
  const float* hs = h   + ((size_t)s * Hh + hd) * Cc;
  float*       ao = acc + ((size_t)d * Hh + hd) * Cc;
  for (int c0 = lane * 4; c0 < Cc; c0 += 128) {
    float4 v = *reinterpret_cast<const float4*>(hs + c0);  // global_load_b128
    atomicAdd(&ao[c0 + 0], coef * v.x);
    atomicAdd(&ao[c0 + 1], coef * v.y);
    atomicAdd(&ao[c0 + 2], coef * v.z);
    atomicAdd(&ao[c0 + 3], coef * v.w);
  }
}

// ---------------------------------------------------------------- epilogues
__global__ void bias_elu_f16(const float* __restrict__ acc, const float* __restrict__ b,
                             _Float16* __restrict__ out, int n, int HC) {
  int i = blockIdx.x * blockDim.x + threadIdx.x;
  if (i >= n) return;
  float v = acc[i] + b[i % HC];
  v = v > 0.f ? v : expm1f(v);  // elu
  out[i] = (_Float16)v;
}

__global__ void bias_elu_f32(const float* __restrict__ acc, const float* __restrict__ b,
                             float* __restrict__ out, int n, int HC) {
  int i = blockIdx.x * blockDim.x + threadIdx.x;
  if (i >= n) return;
  float v = acc[i] + b[i % HC];
  v = v > 0.f ? v : expm1f(v);
  out[i] = v;
}

// ---------------------------------------------------------------- host driver
extern "C" void kernel_launch(void* const* d_in, const int* in_sizes, int n_in,
                              void* d_out, int out_size, void* d_ws, size_t ws_size,
                              hipStream_t stream) {
  const float*     x   = (const float*)d_in[0];
  const long long* ei  = (const long long*)d_in[1];  // int64 edge_index [2,E]
  const long long* src = ei;
  const long long* dst = ei + NEDGES;
  const float *W1 = (const float*)d_in[2],  *as1 = (const float*)d_in[3],
              *ad1 = (const float*)d_in[4], *b1  = (const float*)d_in[5];
  const float *W2 = (const float*)d_in[6],  *as2 = (const float*)d_in[7],
              *ad2 = (const float*)d_in[8], *b2  = (const float*)d_in[9];
  const float *W3 = (const float*)d_in[10], *as3 = (const float*)d_in[11],
              *ad3 = (const float*)d_in[12], *b3 = (const float*)d_in[13];

  // ---- workspace carve (256B aligned)
  char* wp = (char*)d_ws;
  auto carve = [&](size_t bytes) -> void* {
    void* p = (void*)wp;
    wp += (bytes + 255) & ~(size_t)255;
    return p;
  };
  _Float16* a16  = (_Float16*)carve((size_t)NNODES * 512 * 2);  // GEMM A (f16)
  _Float16* w16  = (_Float16*)carve((size_t)512 * 512 * 2);     // W^T (f16)
  float*    hbuf = (float*)carve((size_t)NNODES * 512 * 4);     // GEMM out h
  float*    acc  = (float*)carve((size_t)NNODES * 512 * 4);     // aggregation
  float*    asb  = (float*)carve((size_t)NNODES * 4 * 4);       // alpha_src
  float*    adb  = (float*)carve((size_t)NNODES * 4 * 4);       // alpha_dst
  float*    mbuf = (float*)carve((size_t)NNODES * 4 * 4);       // segment max
  float*    den  = (float*)carve((size_t)NNODES * 4 * 4);       // segment sum
  float*    e_ws = (float*)carve((size_t)NEDGES * 4 * 4);       // edge logits
  float*    p_ws = (float*)carve((size_t)NEDGES * 4 * 4);       // edge exp

  const int T = 256;
  const float NEG_BIG = -3.402823466e38f;

  auto run_layer = [&](const _Float16* a_in, int K, int Hh, int Cc,
                       const float* W, const float* a_s, const float* a_d, const float* bias,
                       _Float16* out16, float* out32) {
    int HC = Hh * Cc;
    conv_w_transpose_f16<<<divup_h(K * HC, T), T, 0, stream>>>(W, w16, K, HC);

    int stripsN = HC / 64;                         // 16x64 strip per wave
    int totalStrips = (NNODES / 16) * stripsN;     // M=50000 divides 16 exactly
    wmma_gemm_f16<<<divup_h(totalStrips * 32, T), T, 0, stream>>>(
        a_in, w16, hbuf, K, HC, stripsN, totalStrips);

    alpha_kernel<<<divup_h(NNODES * Hh * 32, T), T, 0, stream>>>(hbuf, a_s, a_d, asb, adb, Hh, Cc);

    fill_f32<<<divup_h(NNODES * Hh, T), T, 0, stream>>>(mbuf, NEG_BIG, NNODES * Hh);
    fill_f32<<<divup_h(NNODES * Hh, T), T, 0, stream>>>(den, 0.f, NNODES * Hh);
    fill_f32<<<divup_h(NNODES * HC, T), T, 0, stream>>>(acc, 0.f, NNODES * HC);

    edge_logits_max<<<divup_h(NEDGES * Hh, T), T, 0, stream>>>(asb, adb, src, dst, e_ws, mbuf, Hh);
    edge_exp_sum<<<divup_h(NEDGES * Hh, T), T, 0, stream>>>(e_ws, mbuf, dst, p_ws, den, Hh);
    edge_scatter<<<divup_h(NEDGES * Hh * 32, T), T, 0, stream>>>(
        p_ws, den, hbuf, src, dst, acc, Hh, Cc);

    if (out16)
      bias_elu_f16<<<divup_h(NNODES * HC, T), T, 0, stream>>>(acc, bias, out16, NNODES * HC, HC);
    else
      bias_elu_f32<<<divup_h(NNODES * HC, T), T, 0, stream>>>(acc, bias, out32, NNODES * HC, HC);
  };

  // Layer 1: x [N,64] -> a16, then GAT(64 -> 4x128)
  conv_f32_to_f16<<<divup_h(NNODES * 64, T), T, 0, stream>>>(x, a16, NNODES * 64);
  run_layer(a16, 64, 4, 128, W1, as1, ad1, b1, a16, nullptr);
  // Layer 2: GAT(512 -> 4x128)
  run_layer(a16, 512, 4, 128, W2, as2, ad2, b2, a16, nullptr);
  // Layer 3: GAT(512 -> 1x128), fp32 output
  run_layer(a16, 512, 1, 128, W3, as3, ad3, b3, nullptr, (float*)d_out);
}
